// Drug_PNAConv_77137612636202
// MI455X (gfx1250) — matrix operand order repaired and checked by reading, not compiled, not run on hardware
//
#include <hip/hip_runtime.h>
#include <math.h>

// ---------------- problem constants ----------------
#define N_NODES 60000
#define E_EDGES 200000
#define H_DIM   128
#define EC_DIM  16
#define T_TOW   4
#define F_INS   32
#define NTF     (N_NODES * H_DIM)          // N * T * F_IN = 7,680,000
#define AVG_DEG_LOG 1.2698286638083365f    // from the fixed degree histogram

// ---------------- CDNA5 WMMA types ----------------
typedef __bf16 bf16_t;
typedef bf16_t v16bf __attribute__((ext_vector_type(16)));
typedef float  v8f   __attribute__((ext_vector_type(8)));
typedef unsigned short u16;

// f32 -> bf16 (native cvt; backend should select v_cvt*bf16*)
__device__ __forceinline__ u16 bf1(float f) {
  return __builtin_bit_cast(u16, (bf16_t)f);
}
__device__ __forceinline__ unsigned bf2(float lo, float hi) {
  return (unsigned)bf1(lo) | ((unsigned)bf1(hi) << 16);
}

// monotone float <-> uint mapping so integer atomicMin/Max == IEEE min/max
__device__ __forceinline__ unsigned ford(float f) {
  unsigned u = __builtin_bit_cast(unsigned, f);
  return (u & 0x80000000u) ? ~u : (u | 0x80000000u);
}
__device__ __forceinline__ float funord(unsigned u) {
  unsigned v = (u & 0x80000000u) ? (u ^ 0x80000000u) : ~u;
  return __builtin_bit_cast(float, v);
}

__device__ __forceinline__ v8f wmma_bf16(v16bf a, v16bf b, v8f c) {
  return __builtin_amdgcn_wmma_f32_16x16x32_bf16(
      false, a, false, b, (short)0, c, false, false);
}

// fragment = 16 contiguous bf16 (32 B), pre-swizzled
__device__ __forceinline__ v16bf load_frag(const u16* p) {
  return *(const v16bf*)p;
}
// pre-swizzled global B store: [chunk][tile][lane][16]
__device__ __forceinline__ v16bf load_b(const u16* wb, int chunk, int n_tiles,
                                        int tile, int lane) {
  return load_frag(wb + (((chunk * n_tiles) + tile) * 32 + lane) * 16);
}
__device__ __forceinline__ v8f bias_acc(float b) {
  v8f c;
#pragma unroll
  for (int j = 0; j < 8; ++j) c[j] = b;
  return c;
}

// A-layout (16x32 bf16 chunk): element (m,k) lives at
//   lane = m + 16*((k>>3)&1),  j = ((k&16)>>1)|(k&7)
// so an even-k pair (k,k+1) packs into dword j>>1 of one lane.

__device__ __forceinline__ void scat(float* sum, float* sumsq,
                                     unsigned* mnb, unsigned* mxb,
                                     int i, float v) {
  atomicAdd(&sum[i], v);
  atomicAdd(&sumsq[i], v * v);
  unsigned o = ford(v);
  atomicMin(&mnb[i], o);
  atomicMax(&mxb[i], o);
}

// ---------------- kernel: init aggregation workspace ----------------
__global__ __launch_bounds__(256) void fill_kernel(float* sum, float* sumsq,
                                                   unsigned* mnb, unsigned* mxb,
                                                   float* cnt) {
  int i = blockIdx.x * 256 + threadIdx.x;
  if (i < NTF) {
    sum[i] = 0.0f;
    sumsq[i] = 0.0f;
    mnb[i] = 0xFFFFFFFFu;  // ordered +inf
    mxb[i] = 0u;           // ordered -inf
  }
  if (i < N_NODES) cnt[i] = 0.0f;
}

// ---------------- kernel: collapse bond-encoder @ edge-encoder ----------------
__global__ __launch_bounds__(256) void combine_kernel(const float* W_bond,
                                                      const float* b_bond,
                                                      const float* W_ee,
                                                      const float* b_ee,
                                                      float* Wcomb, float* bcomb) {
  int i = blockIdx.x * 256 + threadIdx.x;
  if (i < EC_DIM * F_INS) {
    int k = i >> 5, f = i & 31;
    float acc = 0.0f;
    for (int h = 0; h < H_DIM; ++h) acc += W_bond[k * H_DIM + h] * W_ee[h * F_INS + f];
    Wcomb[i] = acc;
  } else if (i < EC_DIM * F_INS + F_INS) {
    int f = i - EC_DIM * F_INS;
    float acc = b_ee[f];
    for (int h = 0; h < H_DIM; ++h) acc += b_bond[h] * W_ee[h * F_INS + f];
    bcomb[f] = acc;
  }
}

// ---------------- kernel: pre-swizzle f32 weights into bf16 B-fragments -------
// dst[((chunk*n_tiles + tile)*32 + lane)*16 + j] = W[(chunk*32 + hi8 + (j&7) + (j&8?16:0))*ncols + n]
__global__ __launch_bounds__(256) void pack_b_kernel(const float* __restrict__ W,
                                                     u16* __restrict__ dst,
                                                     int ktot, int ncols) {
  int total = ktot * ncols;
  int n_tiles = ncols >> 4;
  for (int idx = blockIdx.x * 256 + threadIdx.x; idx < total;
       idx += gridDim.x * 256) {
    int j = idx & 15;
    int lane = (idx >> 4) & 31;
    int ct = idx >> 9;
    int tile = ct % n_tiles;
    int kcg = ct / n_tiles;
    int k = kcg * 32 + ((lane >= 16) ? 8 : 0) + (j & 7) + ((j & 8) ? 16 : 0);
    int n = tile * 16 + (lane & 15);
    dst[idx] = bf1(W[k * ncols + n]);
  }
}

// ---------------- edge pass ----------------
// 128 thr = 4 waves; tile = 16 edges; wave w = tower w.
__global__ __launch_bounds__(128) void edge_kernel(
    const float* __restrict__ atom_x, const float* __restrict__ bond_x,
    const int* __restrict__ ei, const float* __restrict__ Wcomb,
    const float* __restrict__ bcomb, const u16* __restrict__ wpre1,
    const float* __restrict__ b_pre1, const u16* __restrict__ wpre2,
    const float* __restrict__ b_pre2, float* sum, float* sumsq,
    unsigned* mnb, unsigned* mxb, float* cnt) {
  __shared__ int sdst[16], ssrc[16];
  __shared__ __align__(32) u16 afeat[T_TOW][2][32][16];  // xi=chunk0 xj=chunk1 (bf16, A-layout)
  __shared__ __align__(32) u16 sea_a[32][16];            // ea chunk (shared by towers)
  __shared__ __align__(32) u16 out1a[T_TOW][32][16];     // relu(h1), A-layout

  const int tid = threadIdx.x;
  const int e0 = blockIdx.x * 16;

  if (tid < 16) {
    ssrc[tid] = ei[e0 + tid];            // atom_edge_index[0] = src
    int d = ei[E_EDGES + e0 + tid];      // atom_edge_index[1] = dst
    sdst[tid] = d;
    atomicAdd(&cnt[d], 1.0f);
  }
  __syncthreads();

  // gather xi/xj, convert+swizzle directly into A-fragment layout (pair-packed)
  for (int o = tid; o < 16 * 64; o += 128) {
    int m = o >> 6;
    int cp = (o & 63) * 2;               // even column in [0,128)
    int t = cp >> 5, k = cp & 31;
    int lane_a = m + 16 * ((k >> 3) & 1);
    int dw = (((k & 16) >> 1) | (k & 7)) >> 1;
    float2 vi = *(const float2*)&atom_x[(long)sdst[m] * H_DIM + cp];
    float2 vj = *(const float2*)&atom_x[(long)ssrc[m] * H_DIM + cp];
    ((unsigned*)&afeat[t][0][lane_a][0])[dw] = bf2(vi.x, vi.y);
    ((unsigned*)&afeat[t][1][lane_a][0])[dw] = bf2(vj.x, vj.y);
  }
  // fused edge encoder (collapsed K=16 weight), pair-packed into A-layout
  for (int o = tid; o < 16 * 16; o += 128) {
    int el = o >> 4;
    int f = (o & 15) * 2;
    float a0 = bcomb[f], a1 = bcomb[f + 1];
    const float* bx = &bond_x[(long)(e0 + el) * EC_DIM];
#pragma unroll
    for (int k = 0; k < EC_DIM; ++k) {
      float w = bx[k];
      a0 += w * Wcomb[k * F_INS + f];
      a1 += w * Wcomb[k * F_INS + f + 1];
    }
    int lane_a = el + 16 * ((f >> 3) & 1);
    int dw = (((f & 16) >> 1) | (f & 7)) >> 1;
    ((unsigned*)&sea_a[lane_a][0])[dw] = bf2(a0, a1);
  }
  __syncthreads();

  const int wv = tid >> 5, lane = tid & 31, t = wv;
  const int mrow = (lane >= 16) ? 8 : 0, ncol = lane & 15;

  // pre layer 1: [xi_t | xj_t | ea] (16x96) @ W_pre1[t] + b -> relu
  v8f c0 = bias_acc(b_pre1[t * F_INS + ncol]);
  v8f c1 = bias_acc(b_pre1[t * F_INS + 16 + ncol]);
#pragma unroll
  for (int kc = 0; kc < 3; ++kc) {
    const u16* ap = (kc < 2) ? &afeat[t][kc][0][0] : &sea_a[0][0];
    v16bf a = load_frag(ap + lane * 16);
    c0 = wmma_bf16(a, load_b(wpre1, t * 3 + kc, 2, 0, lane), c0);
    c1 = wmma_bf16(a, load_b(wpre1, t * 3 + kc, 2, 1, lane), c1);
  }
  // relu + restage into A-layout for layer 2
  {
    int half = (ncol >> 3) & 1;
    int jj0 = ncol & 7, jj1 = 8 + (ncol & 7);
#pragma unroll
    for (int j = 0; j < 8; ++j) {
      int la = (j + mrow) + 16 * half;
      out1a[wv][la][jj0] = bf1(fmaxf(c0[j], 0.0f));
      out1a[wv][la][jj1] = bf1(fmaxf(c1[j], 0.0f));
    }
  }
  __syncthreads();

  // pre layer 2: 32x32
  v8f d0 = bias_acc(b_pre2[t * F_INS + ncol]);
  v8f d1 = bias_acc(b_pre2[t * F_INS + 16 + ncol]);
  {
    v16bf a2 = load_frag(&out1a[wv][lane][0]);
    d0 = wmma_bf16(a2, load_b(wpre2, t, 2, 0, lane), d0);
    d1 = wmma_bf16(a2, load_b(wpre2, t, 2, 1, lane), d1);
  }

  // multi-aggregation scatter into L2-resident state
#pragma unroll
  for (int j = 0; j < 8; ++j) {
    int m = j + mrow;
    int d = sdst[m];
    int base = (d * T_TOW + t) * F_INS;
    scat(sum, sumsq, mnb, mxb, base + ncol, d0[j]);
    scat(sum, sumsq, mnb, mxb, base + ncol + 16, d1[j]);
  }
}

// ---------------- node pass ----------------
// 128 thr = 4 waves; tile = 16 nodes; wave w = tower w, then joint final linear.
// Degree scalers applied to f32 accumulators: [agg|agg*s1|agg*s2]@W
//   = agg@W0 + diag(s1)(agg@W1) + diag(s2)(agg@W2)
__global__ __launch_bounds__(128) void node_kernel(
    const float* __restrict__ atom_x, const float* __restrict__ sum,
    const float* __restrict__ sumsq, const unsigned* __restrict__ mnb,
    const unsigned* __restrict__ mxb, const float* __restrict__ cnt,
    const u16* __restrict__ wpost1, const float* __restrict__ b_post1,
    const u16* __restrict__ wpost2, const float* __restrict__ b_post2,
    const u16* __restrict__ wlin, const float* __restrict__ b_lin,
    const float* __restrict__ ln_w, const float* __restrict__ ln_b,
    float* __restrict__ out) {
  __shared__ float xtile[16][H_DIM];                       // residual (8 KB)
  __shared__ __align__(32) u16 afeat[T_TOW][6][32][16];    // x + 5 agg chunks (24 KB)
  __shared__ __align__(32) u16 out1a[T_TOW][32][16];       // 4 KB
  __shared__ __align__(32) u16 out2a[T_TOW][32][16];       // final-GEMM A (4 KB)
  __shared__ float finalbuf[16][H_DIM];                    // 8 KB
  __shared__ float sdeg[16], ssc1[16], ssc2[16], smu[16], srstd[16];
  __shared__ int shas[16];

  const int tid = threadIdx.x;
  const int n0 = blockIdx.x * 16;

  if (tid < 16) {
    float c = cnt[n0 + tid];
    float dg = fmaxf(c, 1.0f);
    sdeg[tid] = dg;
    float ld = logf(dg + 1.0f);
    ssc1[tid] = ld * (1.0f / AVG_DEG_LOG);
    ssc2[tid] = AVG_DEG_LOG / ld;
    shas[tid] = (c > 0.0f);
  }
  __syncthreads();

  // x chunk (A-layout) + f32 copy for residual
  for (int o = tid; o < 16 * 64; o += 128) {
    int m = o >> 6;
    int cp = (o & 63) * 2;
    float2 v = *(const float2*)&atom_x[(long)(n0 + m) * H_DIM + cp];
    xtile[m][cp] = v.x;
    xtile[m][cp + 1] = v.y;
    int t = cp >> 5, k = cp & 31;
    int lane_a = m + 16 * ((k >> 3) & 1);
    int dw = (((k & 16) >> 1) | (k & 7)) >> 1;
    ((unsigned*)&afeat[t][0][lane_a][0])[dw] = bf2(v.x, v.y);
  }
  // finalize [s, mean, min, max, std] -> bf16 A-layout chunks 1..5
  for (int o = tid; o < 16 * T_TOW * 16; o += 128) {
    int n = o >> 6;
    int r = o & 63;
    int t = r >> 4;
    int f = (r & 15) * 2;
    int gi = ((n0 + n) * T_TOW + t) * F_INS + f;
    float2 s2 = *(const float2*)&sum[gi];
    float2 q2 = *(const float2*)&sumsq[gi];
    uint2 mn2 = *(const uint2*)&mnb[gi];
    uint2 mx2 = *(const uint2*)&mxb[gi];
    float dg = sdeg[n];
    int hs = shas[n];
    float mean0 = s2.x / dg, mean1 = s2.y / dg;
    float sd0 = sqrtf(fmaxf(q2.x / dg - mean0 * mean0, 0.0f) + 1e-5f);
    float sd1 = sqrtf(fmaxf(q2.y / dg - mean1 * mean1, 0.0f) + 1e-5f);
    float mn0 = hs ? funord(mn2.x) : 0.0f, mn1 = hs ? funord(mn2.y) : 0.0f;
    float mx0 = hs ? funord(mx2.x) : 0.0f, mx1 = hs ? funord(mx2.y) : 0.0f;
    int lane_a = n + 16 * ((f >> 3) & 1);
    int dw = (((f & 16) >> 1) | (f & 7)) >> 1;
    ((unsigned*)&afeat[t][1][lane_a][0])[dw] = bf2(s2.x, s2.y);
    ((unsigned*)&afeat[t][2][lane_a][0])[dw] = bf2(mean0, mean1);
    ((unsigned*)&afeat[t][3][lane_a][0])[dw] = bf2(mn0, mn1);
    ((unsigned*)&afeat[t][4][lane_a][0])[dw] = bf2(mx0, mx1);
    ((unsigned*)&afeat[t][5][lane_a][0])[dw] = bf2(sd0, sd1);
  }
  __syncthreads();

  const int wv = tid >> 5, lane = tid & 31, t = wv;
  const int mrow = (lane >= 16) ? 8 : 0, ncol = lane & 15;

  // post layer 1 with scaler trick (W_post1[t] rows: 0-31 x, 32-191 agg,
  // 192-351 agg*sc1, 352-511 agg*sc2 => chunks 0 | 1-5 | 6-10 | 11-15)
  v8f p00 = bias_acc(b_post1[t * F_INS + ncol]);
  v8f p01 = bias_acc(b_post1[t * F_INS + 16 + ncol]);
  v8f p10 = bias_acc(0.0f), p11 = bias_acc(0.0f);
  v8f p20 = bias_acc(0.0f), p21 = bias_acc(0.0f);
  {
    v16bf a = load_frag(&afeat[t][0][lane][0]);
    p00 = wmma_bf16(a, load_b(wpost1, t * 16 + 0, 2, 0, lane), p00);
    p01 = wmma_bf16(a, load_b(wpost1, t * 16 + 0, 2, 1, lane), p01);
  }
#pragma unroll
  for (int s = 0; s < 5; ++s) {
    v16bf a = load_frag(&afeat[t][1 + s][lane][0]);
    p00 = wmma_bf16(a, load_b(wpost1, t * 16 + 1 + s, 2, 0, lane), p00);
    p01 = wmma_bf16(a, load_b(wpost1, t * 16 + 1 + s, 2, 1, lane), p01);
    p10 = wmma_bf16(a, load_b(wpost1, t * 16 + 6 + s, 2, 0, lane), p10);
    p11 = wmma_bf16(a, load_b(wpost1, t * 16 + 6 + s, 2, 1, lane), p11);
    p20 = wmma_bf16(a, load_b(wpost1, t * 16 + 11 + s, 2, 0, lane), p20);
    p21 = wmma_bf16(a, load_b(wpost1, t * 16 + 11 + s, 2, 1, lane), p21);
  }
  // combine partials with per-row scalers, relu, restage for layer 2
  {
    int half = (ncol >> 3) & 1;
    int jj0 = ncol & 7, jj1 = 8 + (ncol & 7);
#pragma unroll
    for (int j = 0; j < 8; ++j) {
      int m = j + mrow;
      float s1 = ssc1[m], s2 = ssc2[m];
      float v0 = p00[j] + s1 * p10[j] + s2 * p20[j];
      float v1 = p01[j] + s1 * p11[j] + s2 * p21[j];
      int la = m + 16 * half;
      out1a[wv][la][jj0] = bf1(fmaxf(v0, 0.0f));
      out1a[wv][la][jj1] = bf1(fmaxf(v1, 0.0f));
    }
  }
  __syncthreads();

  // post layer 2: 32x32 -> out2 staged as final-GEMM A chunk t
  {
    v8f d0 = bias_acc(b_post2[t * F_INS + ncol]);
    v8f d1 = bias_acc(b_post2[t * F_INS + 16 + ncol]);
    v16bf a2 = load_frag(&out1a[wv][lane][0]);
    d0 = wmma_bf16(a2, load_b(wpost2, t, 2, 0, lane), d0);
    d1 = wmma_bf16(a2, load_b(wpost2, t, 2, 1, lane), d1);
    int half = (ncol >> 3) & 1;
    int jj0 = ncol & 7, jj1 = 8 + (ncol & 7);
#pragma unroll
    for (int j = 0; j < 8; ++j) {
      int la = (j + mrow) + 16 * half;
      out2a[t][la][jj0] = bf1(d0[j]);
      out2a[t][la][jj1] = bf1(d1[j]);
    }
  }
  __syncthreads();

  // final linear: out2[16x128] @ W_lin; wave w owns columns w*32..w*32+31
  {
    v8f e0 = bias_acc(b_lin[wv * 32 + ncol]);
    v8f e1 = bias_acc(b_lin[wv * 32 + 16 + ncol]);
#pragma unroll
    for (int kc = 0; kc < 4; ++kc) {
      v16bf a = load_frag(&out2a[kc][lane][0]);
      e0 = wmma_bf16(a, load_b(wlin, kc, 8, 2 * wv, lane), e0);
      e1 = wmma_bf16(a, load_b(wlin, kc, 8, 2 * wv + 1, lane), e1);
    }
#pragma unroll
    for (int j = 0; j < 8; ++j) {
      finalbuf[j + mrow][wv * 32 + ncol] = e0[j];
      finalbuf[j + mrow][wv * 32 + 16 + ncol] = e1[j];
    }
  }
  __syncthreads();

  // LayerNorm statistics (one lane per node)
  if (tid < 16) {
    float mu = 0.0f;
    for (int c = 0; c < H_DIM; ++c) mu += finalbuf[tid][c];
    mu *= (1.0f / H_DIM);
    float var = 0.0f;
    for (int c = 0; c < H_DIM; ++c) {
      float dv = finalbuf[tid][c] - mu;
      var += dv * dv;
    }
    var *= (1.0f / H_DIM);
    smu[tid] = mu;
    srstd[tid] = rsqrtf(var + 1e-5f);
  }
  __syncthreads();

  // y = LN(out)*ln_w + ln_b ; result = atom_x + relu(y)
  for (int o = tid; o < 16 * H_DIM; o += 128) {
    int n = o >> 7, c = o & 127;
    float y = (finalbuf[n][c] - smu[n]) * srstd[n] * ln_w[c] + ln_b[c];
    out[(long)(n0 + n) * H_DIM + c] = xtile[n][c] + fmaxf(y, 0.0f);
  }
}

// ---------------- host-side launcher ----------------
extern "C" void kernel_launch(void* const* d_in, const int* in_sizes, int n_in,
                              void* d_out, int out_size, void* d_ws, size_t ws_size,
                              hipStream_t stream) {
  const float* atom_x  = (const float*)d_in[0];
  const float* bond_x  = (const float*)d_in[1];
  const int*   ei      = (const int*)d_in[2];
  const float* W_bond  = (const float*)d_in[3];
  const float* b_bond  = (const float*)d_in[4];
  const float* W_ee    = (const float*)d_in[5];
  const float* b_ee    = (const float*)d_in[6];
  const float* W_pre1  = (const float*)d_in[7];
  const float* b_pre1  = (const float*)d_in[8];
  const float* W_pre2  = (const float*)d_in[9];
  const float* b_pre2  = (const float*)d_in[10];
  const float* W_post1 = (const float*)d_in[11];
  const float* b_post1 = (const float*)d_in[12];
  const float* W_post2 = (const float*)d_in[13];
  const float* b_post2 = (const float*)d_in[14];
  const float* W_lin   = (const float*)d_in[15];
  const float* b_lin   = (const float*)d_in[16];
  const float* ln_w    = (const float*)d_in[17];
  const float* ln_b    = (const float*)d_in[18];
  float* out = (float*)d_out;

  // workspace: L2-resident aggregation state + fused/pre-swizzled weights
  char* ws = (char*)d_ws;
  size_t off = 0;
  float*    sumb  = (float*)(ws + off);    off += (size_t)NTF * 4;
  float*    sumsq = (float*)(ws + off);    off += (size_t)NTF * 4;
  unsigned* mnb   = (unsigned*)(ws + off); off += (size_t)NTF * 4;
  unsigned* mxb   = (unsigned*)(ws + off); off += (size_t)NTF * 4;
  float*    cntb  = (float*)(ws + off);    off += (size_t)N_NODES * 4;
  float*    Wcomb = (float*)(ws + off);    off += (size_t)EC_DIM * F_INS * 4;
  float*    bcomb = (float*)(ws + off);    off += (size_t)F_INS * 4;
  u16* wpre1_bf  = (u16*)(ws + off); off += (size_t)(T_TOW * 96) * F_INS * 2;
  u16* wpre2_bf  = (u16*)(ws + off); off += (size_t)(T_TOW * 32) * F_INS * 2;
  u16* wpost1_bf = (u16*)(ws + off); off += (size_t)(T_TOW * 512) * F_INS * 2;
  u16* wpost2_bf = (u16*)(ws + off); off += (size_t)(T_TOW * 32) * F_INS * 2;
  u16* wlin_bf   = (u16*)(ws + off); off += (size_t)H_DIM * H_DIM * 2;
  (void)ws_size; (void)in_sizes; (void)n_in; (void)out_size;

  fill_kernel<<<(NTF + 255) / 256, 256, 0, stream>>>(sumb, sumsq, mnb, mxb, cntb);
  combine_kernel<<<3, 256, 0, stream>>>(W_bond, b_bond, W_ee, b_ee, Wcomb, bcomb);
  pack_b_kernel<<<48, 256, 0, stream>>>(W_pre1, wpre1_bf, T_TOW * 96, F_INS);
  pack_b_kernel<<<16, 256, 0, stream>>>(W_pre2, wpre2_bf, T_TOW * 32, F_INS);
  pack_b_kernel<<<256, 256, 0, stream>>>(W_post1, wpost1_bf, T_TOW * 512, F_INS);
  pack_b_kernel<<<16, 256, 0, stream>>>(W_post2, wpost2_bf, T_TOW * 32, F_INS);
  pack_b_kernel<<<64, 256, 0, stream>>>(W_lin, wlin_bf, H_DIM, H_DIM);

  edge_kernel<<<E_EDGES / 16, 128, 0, stream>>>(
      atom_x, bond_x, ei, Wcomb, bcomb, wpre1_bf, b_pre1, wpre2_bf, b_pre2,
      sumb, sumsq, mnb, mxb, cntb);
  node_kernel<<<N_NODES / 16, 128, 0, stream>>>(
      atom_x, sumb, sumsq, mnb, mxb, cntb, wpost1_bf, b_post1, wpost2_bf,
      b_post2, wlin_bf, b_lin, ln_w, ln_b, out);
}
